// InputTransition_76441827934577
// MI455X (gfx1250) — matrix operand order repaired and checked by reading, not compiled, not run on hardware
//
#include <hip/hip_runtime.h>
#include <math.h>

// ---------------------------------------------------------------------------
// SubMConv3d (rulebook gather + GEMM) + SparseBN + residual + ELU for MI455X.
//
// Roofline: nbr table = 500 MB streamed once dominates (HBM floor ~30 us at
// 23.3 TB/s incl. the 64 MB d_out staging re-read). Compute (2 GMAC) folds
// into 4x v_wmma_f32_16x16x32_bf16 per 16-voxel tile -> matrix units make
// arithmetic free. nbr loads are non-temporal so the 192 MB L2 stays
// reserved for feats (4 MB, reused 125x) and the staged pre-BN output.
// Training-mode BN makes the conv bias `b` cancel exactly -> skipped.
// ---------------------------------------------------------------------------

#define KOFF   125     // 5x5x5 kernel offsets
#define FCH    16      // output channels
#define BN_EPS 1e-5f

typedef __attribute__((ext_vector_type(16))) __bf16 v16bf;
typedef __attribute__((ext_vector_type(8)))  float  v8f;
typedef __attribute__((ext_vector_type(4)))  float  v4f;
typedef int int2v __attribute__((ext_vector_type(2), aligned(4)));

// ws layout (floats):
//   [0..15]  sum[f]      [16..31] sumsq[f]
//   [32..47] scale[f]    [48..63] shift[f]
//   [64.. )  B-matrix WMMA fragments: 4 k-steps * 32 lanes * 16 bf16 (4 KB)
#define WS_SUM    0
#define WS_SQ     16
#define WS_SCALE  32
#define WS_SHIFT  48
#define WS_BFRAG  64

// --- Pass 0: zero BN accumulators, prebuild bf16 B fragments -----------------
// 16-bit B 32x16 striping used here (contiguous k): lane holds column f = lane&15; lane-half h
// selects K range: element e of the v16bf is K = 32*step + 16*h + e.
__global__ void prep_kernel(const float* __restrict__ W, float* __restrict__ ws) {
    int t = threadIdx.x;
    if (t < 64) ws[t] = 0.f;
    if (t < 32) {
        __bf16* bf = (__bf16*)(ws + WS_BFRAG);
        int h = t >> 4, f = t & 15;
        for (int s = 0; s < 4; ++s)
            for (int e = 0; e < 16; ++e) {
                int kk = 32 * s + 16 * h + e;
                float w = (kk < KOFF) ? W[kk * FCH + f] : 0.f;   // W[k][0][f]
                bf[(s * 32 + t) * 16 + e] = (__bf16)w;
            }
    }
}

// --- Pass 1: gather + WMMA GEMM, write pre-BN to d_out, accumulate stats -----
__global__ __launch_bounds__(256) void spconv_wmma_kernel(
    const float* __restrict__ feats, const int* __restrict__ nbr,
    float* __restrict__ out_pre, float* ws, int N, int tiles)
{
    __shared__ float sh_sum[FCH], sh_sq[FCH];
    if (threadIdx.x < FCH) { sh_sum[threadIdx.x] = 0.f; sh_sq[threadIdx.x] = 0.f; }
    __syncthreads();

    int lane = threadIdx.x & 31;
    int wave = threadIdx.x >> 5;
    // force tile wave-uniform so the guard is a scalar branch (EXEC stays all-1s
    // around the WMMAs, as the ISA requires)
    int tile = __builtin_amdgcn_readfirstlane((int)(blockIdx.x * 8 + wave));

    if (tile < tiles) {
        int h  = lane >> 4;       // lane half
        int m  = lane & 15;       // A-matrix row / C column
        int n0 = tile * 16;
        bool rowok = (n0 + m) < N;

        const v16bf* Bf = (const v16bf*)(ws + WS_BFRAG);
        v16bf bfr[4];
        #pragma unroll
        for (int s = 0; s < 4; ++s) bfr[s] = Bf[s * 32 + lane];

        const int* nrow = nbr + (size_t)(n0 + m) * KOFF;

        v8f acc = {};
        #pragma unroll
        for (int s = 0; s < 4; ++s) {
            // A striping matching the B fragments: element e of the v16bf is
            // K = 32*s + 16*h + e.  Consecutive
            // (even,odd) element pairs read adjacent kk -> one NT b64 load.
            v16bf a;
            #pragma unroll
            for (int p = 0; p < 8; ++p) {
                int e  = 2 * p;
                int kk = 32 * s + 16 * h + e;
                float g0 = 0.f, g1 = 0.f;
                if (rowok) {
                    if (kk + 1 < KOFF) {          // always true for s < 3
                        int2v pr = __builtin_nontemporal_load(
                            (const int2v*)(nrow + kk));
                        if (pr.x < N) g0 = feats[pr.x];
                        if (pr.y < N) g1 = feats[pr.y];
                    } else if (kk < KOFF) {       // kk == 124 tail (s==3, h==1)
                        int idx = __builtin_nontemporal_load(nrow + kk);
                        if (idx < N) g0 = feats[idx];
                    }
                }
                a[e]     = (__bf16)g0;
                a[e + 1] = (__bf16)g1;
            }
            acc = __builtin_amdgcn_wmma_f32_16x16x32_bf16(
                false, a, false, bfr[s], (short)0, acc, false, false);
        }

        // C layout: acc[r] = out[n0 + r + 8h][f=m] -> coalesced 64B segments
        float s1 = 0.f, s2 = 0.f;
        #pragma unroll
        for (int r = 0; r < 8; ++r) {
            float v = acc[r];
            int   n = n0 + 8 * h + r;
            if (n < N) out_pre[(size_t)n * FCH + m] = v;
            s1 += v; s2 += v * v;                 // invalid rows contribute 0
        }
        // fold the two lane-halves (same channel m, different rows)
        s1 += __shfl_xor(s1, 16, 32);
        s2 += __shfl_xor(s2, 16, 32);
        if (lane < FCH) { atomicAdd(&sh_sum[lane], s1); atomicAdd(&sh_sq[lane], s2); }
    }
    __syncthreads();
    if (threadIdx.x < FCH)
        atomicAdd(&ws[WS_SUM + threadIdx.x], sh_sum[threadIdx.x]);
    else if (threadIdx.x < 2 * FCH)
        atomicAdd(&ws[WS_SQ + threadIdx.x - FCH], sh_sq[threadIdx.x - FCH]);
}

// --- Pass 2: finalize biased-variance BN affine ------------------------------
__global__ void stats_kernel(const float* __restrict__ gamma,
                             const float* __restrict__ beta,
                             float* ws, int N) {
    int f = threadIdx.x;
    if (f < FCH) {
        float inv  = 1.f / (float)N;
        float mean = ws[WS_SUM + f] * inv;
        float var  = ws[WS_SQ + f] * inv - mean * mean;
        float sc   = gamma[f] * rsqrtf(var + BN_EPS);
        ws[WS_SCALE + f] = sc;
        ws[WS_SHIFT + f] = beta[f] - mean * sc;
    }
}

// --- Pass 3: in-place BN apply + tiled residual + ELU (float4 per thread) ----
__global__ __launch_bounds__(256) void finalize_kernel(
    const float* __restrict__ feats, const float* __restrict__ ws,
    float* __restrict__ out, int nvox)
{
    int stride = gridDim.x * blockDim.x;   // multiple of 4 -> quad id invariant
    int quads  = nvox * 4;                 // 4 channel-quads per voxel
    int q0     = blockIdx.x * blockDim.x + threadIdx.x;
    int f0     = (q0 & 3) * 4;
    v4f sc = *(const v4f*)(ws + WS_SCALE + f0);
    v4f sh = *(const v4f*)(ws + WS_SHIFT + f0);
    for (int q = q0; q < quads; q += stride) {
        int n = q >> 2;
        v4f v = *(v4f*)(out + (size_t)q * 4);
        float fn = feats[n];               // CIN==1: channel tile == broadcast
        #pragma unroll
        for (int j = 0; j < 4; ++j) {
            float y = v[j] * sc[j] + sh[j] + fn;
            v[j] = y > 0.f ? y : expm1f(y);          // ELU
        }
        *(v4f*)(out + (size_t)q * 4) = v;
    }
}

extern "C" void kernel_launch(void* const* d_in, const int* in_sizes, int n_in,
                              void* d_out, int out_size, void* d_ws, size_t ws_size,
                              hipStream_t stream) {
    const float* feats = (const float*)d_in[0];
    const float* W     = (const float*)d_in[1];
    // d_in[2] = conv bias b: cancels exactly under training-mode batchnorm.
    const float* gamma = (const float*)d_in[3];
    const float* beta  = (const float*)d_in[4];
    const int*   nbr   = (const int*)d_in[5];
    float* out = (float*)d_out;
    float* ws  = (float*)d_ws;

    int N       = in_sizes[0];             // CIN == 1 -> feats is [N]
    int tiles   = (N + 15) / 16;
    int blocks1 = (tiles + 7) / 8;         // 8 waves (tiles) per 256-thr block

    prep_kernel<<<1, 64, 0, stream>>>(W, ws);
    spconv_wmma_kernel<<<blocks1, 256, 0, stream>>>(feats, nbr, out, ws, N, tiles);
    stats_kernel<<<1, 32, 0, stream>>>(gamma, beta, ws, N);

    int quads   = N * 4;
    int blocks3 = (quads + 255) / 256;
    if (blocks3 > 16384) blocks3 = 16384;  // grid-stride covers the rest
    finalize_kernel<<<blocks3, 256, 0, stream>>>(feats, ws, out, N);
}
